// OriLinearGNN_6846177869862
// MI455X (gfx1250) — matrix operand order repaired and checked by reading, not compiled, not run on hardware
//
#include <hip/hip_runtime.h>
#include <hip/hip_bf16.h>
#include <math.h>
#include <stdint.h>

typedef __attribute__((ext_vector_type(16))) __bf16 v16bf;
typedef __attribute__((ext_vector_type(8)))  float  v8f;
typedef __attribute__((ext_vector_type(4)))  float  f32x4;
typedef __attribute__((ext_vector_type(4)))  uint32_t u32x4;
typedef __attribute__((ext_vector_type(8)))  int32_t  i32x8;
typedef __attribute__((ext_vector_type(4)))  int32_t  i32x4;

#define MU_OVER_S 0.05625f   // 0.9 / 16

#if __has_builtin(__builtin_amdgcn_tanhf)
#define TANH(x) __builtin_amdgcn_tanhf(x)
#elif __has_builtin(__builtin_amdgcn_tanh_f32)
#define TANH(x) __builtin_amdgcn_tanh_f32(x)
#else
#define TANH(x) tanhf(x)
#endif

#if __has_builtin(__builtin_amdgcn_tensor_load_to_lds) && \
    __has_builtin(__builtin_amdgcn_s_wait_tensorcnt)
#define USE_TDM 1
#else
#define USE_TDM 0
#endif

// ---------------------------------------------------------------------------
// K0: convert xi_w [256,256] and rou_w [128,16] (f32 row-major) into bf16
// WMMA B-fragment layout.
//   B frag(n_blk,kk): lane L holds column j = n_blk*16 + (L&15),
//                     K(p) = kk*32 + 16*(L>>4) + p
// ---------------------------------------------------------------------------
__global__ void prep_weights(const float* __restrict__ xi_w,
                             const float* __restrict__ rou_w,
                             __bf16* __restrict__ Bw,    // [16*8][32] v16bf
                             __bf16* __restrict__ Brou)  // [4][32] v16bf
{
    int tid  = blockIdx.x * blockDim.x + threadIdx.x;
    int lane = tid & 31;
    int frag = tid >> 5;
    int half = lane >> 4, lm = lane & 15;
    if (frag < 16 * 8) {
        int n_blk = frag >> 3, kk = frag & 7;
        int j = n_blk * 16 + lm;
        v16bf o;
#pragma unroll
        for (int p = 0; p < 16; ++p) {
            int K = kk * 32 + half * 16 + p;
            o[p] = (__bf16)xi_w[K * 256 + j];
        }
        *((v16bf*)Bw + (size_t)frag * 32 + lane) = o;
    } else if (frag < 16 * 8 + 4) {
        int kk = frag - 128;
        v16bf o;
#pragma unroll
        for (int p = 0; p < 16; ++p) {
            int K = kk * 32 + half * 16 + p;
            o[p] = (__bf16)rou_w[K * 16 + lm];
        }
        *((v16bf*)Brou + (size_t)kk * 32 + lane) = o;
    }
}

// ---------------------------------------------------------------------------
// K1: one wave per 16-edge tile.
//  - gathers emb rows for src/nbr, converts to bf16 into A-fragment layout Xb
//      A frag(kk): lane L holds row M = L&15 (edge-in-tile),
//                  K(p) = kk*32 + 16*(L>>4) + p
//  - computes b = tanh(Xn @ rou_w + rou_b) with 4 WMMAs (K=128)
//  - computes scale[e] = (mu/S)/dg[e]
// ---------------------------------------------------------------------------
__global__ void __launch_bounds__(256) prep_edges(
    const int*   __restrict__ X_Node,
    const int*   __restrict__ X_Neis,
    const int*   __restrict__ dg,
    const float* __restrict__ emb,
    const __bf16* __restrict__ Brou,
    const float* __restrict__ rou_b,
    __bf16* __restrict__ Xb,      // [ntiles][8][32] v16bf
    float*  __restrict__ bvec,    // [E][16]
    float*  __restrict__ scale,   // [E]
    int ntiles)
{
    int lane  = threadIdx.x & 31;
    int wave  = (blockIdx.x * blockDim.x + threadIdx.x) >> 5;
    int nwav  = (gridDim.x * blockDim.x) >> 5;
    int half  = lane >> 4, lm = lane & 15;

    v16bf brf[4];
#pragma unroll
    for (int kk = 0; kk < 4; ++kk)
        brf[kk] = *((const v16bf*)Brou + (size_t)kk * 32 + lane);

    for (int t = wave; t < ntiles; t += nwav) {
        int e    = t * 16 + lm;          // this lane's edge (row M = lm)
        int node = X_Node[e];
        int nei  = X_Neis[e];
        const float* rowN = emb + (size_t)node * 128;
        const float* rowB = emb + (size_t)nei  * 128;
        v16bf a[8];
#pragma unroll
        for (int kk = 0; kk < 8; ++kk) {
            const float* src = (kk < 4) ? rowN : rowB;
            int base = (kk & 3) * 32 + half * 16;  // run: base..+15
            f32x4 q0 = *(const f32x4*)(src + base);
            f32x4 q1 = *(const f32x4*)(src + base + 4);
            f32x4 q2 = *(const f32x4*)(src + base + 8);
            f32x4 q3 = *(const f32x4*)(src + base + 12);
            v16bf o;
#pragma unroll
            for (int i = 0; i < 4; ++i) {
                o[i]      = (__bf16)q0[i];
                o[4 + i]  = (__bf16)q1[i];
                o[8 + i]  = (__bf16)q2[i];
                o[12 + i] = (__bf16)q3[i];
            }
            a[kk] = o;
            *((v16bf*)Xb + ((size_t)t * 8 + kk) * 32 + lane) = o;
        }
        // b = tanh(Xn @ rou_w + rou_b): K = 128 -> 4 WMMAs (node half only)
        v8f c = {};
#pragma unroll
        for (int kk = 0; kk < 4; ++kk)
            c = __builtin_amdgcn_wmma_f32_16x16x32_bf16(
                    false, a[kk], false, brf[kk], (short)0, c, false, false);
        float rb = rou_b[lm];
#pragma unroll
        for (int r = 0; r < 8; ++r) {
            int er = t * 16 + r + 8 * half;        // C row M = r + 8*half
            bvec[(size_t)er * 16 + lm] = TANH(c[r] + rb);
        }
        if (half == 0) scale[e] = MU_OVER_S / (float)dg[e];
    }
}

// ---------------------------------------------------------------------------
// K2: one propagation step. One wave per 16-edge tile; xi_w B-fragments
// staged in 128 KB LDS per block (TDM tensor_load_to_lds when available).
// Per tile and output row-block nb: 8 WMMAs -> tanh -> *scale -> *H(gather),
// register-halving cross-lane reduce over t, +b, atomic scatter.
// first==1: states_old == 0 -> output is just segment_sum(b).
// ---------------------------------------------------------------------------
__global__ void __launch_bounds__(256) prop_step(
    const float* __restrict__ states_old,   // [V][16]
    float*       __restrict__ states_new,   // [V][16] (pre-zeroed)
    const __bf16* __restrict__ Xb,
    const __bf16* __restrict__ Bw,
    const float* __restrict__ xi_b,
    const float* __restrict__ bvec,
    const float* __restrict__ scale,
    const int*   __restrict__ X_Node,
    const int*   __restrict__ X_Neis,
    int ntiles, int first)
{
    extern __shared__ char smem[];
    v16bf* sB = (v16bf*)smem;               // 16*8 frags * 32 lanes = 4096
    int tid  = threadIdx.x;
    int lane = tid & 31;
    int half = lane >> 4, lm = lane & 15;

    if (!first) {
#if USE_TDM
        if (tid < 32) {
            // Tensor DMA: copy 128 KB (32768 f32-sized elements) Bw -> LDS.
            uint32_t ldsOff = (uint32_t)(uintptr_t)smem;   // LDS aperture: low 32 bits
            uint64_t gaddr  = (uint64_t)(uintptr_t)Bw;
            u32x4 g0 = (u32x4)0;
            g0[0] = 1u;                                    // count = 1
            g0[1] = ldsOff;                                // lds_addr
            g0[2] = (uint32_t)gaddr;                       // global_addr[31:0]
            g0[3] = ((uint32_t)(gaddr >> 32) & 0x01FFFFFFu) | 0x80000000u; // addr[56:32] | type=2
            i32x8 g1 = (i32x8)0;
            g1[0] = (int32_t)(2u << 16);                   // wg_mask=0, data_size=4B
            g1[1] = (int32_t)(0x8000u << 16);              // tensor_dim0 = 32768 (bits 63:48)
            g1[2] = (int32_t)(1u << 16);                   // tensor_dim1 = 1
            g1[3] = (int32_t)(0x8000u << 16);              // tile_dim0 = 32768
            g1[4] = 1;                                     // tile_dim1 = 1
            g1[5] = 32768;                                 // tensor_dim0_stride
            i32x4 g2 = (i32x4)0;
            i32x4 g3 = (i32x4)0;
            i32x8 g4 = (i32x8)0;
            __builtin_amdgcn_tensor_load_to_lds(g0, g1, g2, g3, g4, 0);
            __builtin_amdgcn_s_wait_tensorcnt(0);
        }
#else
        const v16bf* gB = (const v16bf*)Bw;
        for (int i = tid; i < 16 * 8 * 32; i += 256) sB[i] = gB[i];
#endif
        __syncthreads();
    }

    // Tile-invariant per-lane bias: xi_b[nb*16 + lm]
    float biasr[16];
#pragma unroll
    for (int nb = 0; nb < 16; ++nb) biasr[nb] = xi_b[nb * 16 + lm];

    // Reduction ownership: lane (within 16-group) bits 3..1 encode owned row.
    const bool b3 = (lm & 8) != 0, b2 = (lm & 4) != 0, b1 = (lm & 2) != 0;
    const int  rown   = (b3 ? 4 : 0) + (b2 ? 2 : 0) + (b1 ? 1 : 0);
    const bool active = (lm & 1) == 0;

    int wave = (blockIdx.x * 256 + tid) >> 5;
    int nwav = (gridDim.x * 256) >> 5;
    for (int t = wave; t < ntiles; t += nwav) {
        int t2 = t + nwav;
        if (t2 < ntiles)  // prefetch next tile's A-fragments into caches
            __builtin_prefetch((const void*)((const v16bf*)Xb +
                               ((size_t)t2 * 8) * 32 + lane), 0, 1);

        float Hg[8], scl[8];
        v16bf a[8];
        if (!first) {
#pragma unroll
            for (int r = 0; r < 8; ++r) {
                int e  = t * 16 + r + 8 * half;   // edge of C row r in this half
                int nd = X_Node[e];
                Hg[r]  = states_old[(size_t)nd * 16 + lm];  // H[e, t=lm], coalesced
                scl[r] = scale[e];
            }
#pragma unroll
            for (int kk = 0; kk < 8; ++kk)
                a[kk] = *((const v16bf*)Xb + ((size_t)t * 8 + kk) * 32 + lane);
        }
        int e_s = t * 16 + rown + 8 * half;        // edge this lane scatters
        int nei = active ? X_Neis[e_s] : 0;

#pragma unroll 1
        for (int nb = 0; nb < 16; ++nb) {          // output row-block s = nb
            float red = 0.0f;
            if (!first) {
                v8f c = {};
#pragma unroll
                for (int kk = 0; kk < 8; ++kk)
                    c = __builtin_amdgcn_wmma_f32_16x16x32_bf16(
                            false, a[kk], false,
                            sB[(size_t)(nb * 8 + kk) * 32 + lane],
                            (short)0, c, false, false);
                float bias = biasr[nb];
                float p[8];
#pragma unroll
                for (int r = 0; r < 8; ++r)
                    p[r] = TANH(c[r] + bias) * scl[r] * Hg[r];
                // register-halving reduce over 16 lanes (t dimension):
                float k4[4];
#pragma unroll
                for (int i = 0; i < 4; ++i) {
                    float snd = b3 ? p[i] : p[i + 4];
                    float rcv = __shfl_xor(snd, 8, 32);
                    k4[i] = (b3 ? p[i + 4] : p[i]) + rcv;
                }
                float k2[2];
#pragma unroll
                for (int i = 0; i < 2; ++i) {
                    float snd = b2 ? k4[i] : k4[i + 2];
                    float rcv = __shfl_xor(snd, 4, 32);
                    k2[i] = (b2 ? k4[i + 2] : k4[i]) + rcv;
                }
                float snd = b1 ? k2[0] : k2[1];
                float rcv = __shfl_xor(snd, 2, 32);
                float s1  = (b1 ? k2[1] : k2[0]) + rcv;
                s1 += __shfl_xor(s1, 1, 32);
                red = s1;                           // full sum for row `rown`
            }
            if (active) {
                float val = red + bvec[(size_t)e_s * 16 + nb];
                atomicAdd(&states_new[(size_t)nei * 16 + nb], val);
            }
        }
    }
}

__global__ void zero_kernel(float* __restrict__ p, int n) {
    int i = blockIdx.x * blockDim.x + threadIdx.x;
    if (i < n) p[i] = 0.0f;
}

// ---------------------------------------------------------------------------
// K3: logits = [emb, states] @ lin_w + lin_b ; softmax over 3 classes
// ---------------------------------------------------------------------------
__global__ void __launch_bounds__(256) final_kernel(
    const float* __restrict__ emb,
    const float* __restrict__ states,
    const float* __restrict__ lin_w,   // [144][3]
    const float* __restrict__ lin_b,   // [3]
    float* __restrict__ out, int V)
{
    __shared__ float slw[144 * 3];
    __shared__ float slb[3];
    int tid = threadIdx.x;
    for (int i = tid; i < 144 * 3; i += blockDim.x) slw[i] = lin_w[i];
    if (tid < 3) slb[tid] = lin_b[tid];
    __syncthreads();
    int v = blockIdx.x * blockDim.x + tid;
    if (v >= V) return;
    float a0 = slb[0], a1 = slb[1], a2 = slb[2];
    const float* er = emb + (size_t)v * 128;
#pragma unroll 4
    for (int k = 0; k < 128; ++k) {
        float x = er[k];
        a0 = fmaf(x, slw[k * 3 + 0], a0);
        a1 = fmaf(x, slw[k * 3 + 1], a1);
        a2 = fmaf(x, slw[k * 3 + 2], a2);
    }
    const float* sr = states + (size_t)v * 16;
#pragma unroll
    for (int k = 0; k < 16; ++k) {
        float x = sr[k];
        a0 = fmaf(x, slw[(128 + k) * 3 + 0], a0);
        a1 = fmaf(x, slw[(128 + k) * 3 + 1], a1);
        a2 = fmaf(x, slw[(128 + k) * 3 + 2], a2);
    }
    float mx = fmaxf(a0, fmaxf(a1, a2));
    float e0 = __expf(a0 - mx), e1 = __expf(a1 - mx), e2 = __expf(a2 - mx);
    float inv = 1.0f / (e0 + e1 + e2);
    out[(size_t)v * 3 + 0] = e0 * inv;
    out[(size_t)v * 3 + 1] = e1 * inv;
    out[(size_t)v * 3 + 2] = e2 * inv;
}

extern "C" void kernel_launch(void* const* d_in, const int* in_sizes, int n_in,
                              void* d_out, int out_size, void* d_ws, size_t ws_size,
                              hipStream_t stream)
{
    const int*   X_Node = (const int*)d_in[0];
    const int*   X_Neis = (const int*)d_in[1];
    const int*   dg     = (const int*)d_in[2];
    const float* emb    = (const float*)d_in[3];
    const float* xi_w   = (const float*)d_in[4];
    const float* xi_b   = (const float*)d_in[5];
    const float* rou_w  = (const float*)d_in[6];
    const float* rou_b  = (const float*)d_in[7];
    const float* lin_w  = (const float*)d_in[8];
    const float* lin_b  = (const float*)d_in[9];

    const int E      = in_sizes[0];
    const int V      = in_sizes[3] / 128;
    const int ntiles = E / 16;               // E = 500000 -> 31250 (exact)

    char* ws = (char*)d_ws;
    size_t off = 0;
    auto take = [&](size_t bytes) {
        size_t r = off;
        off += (bytes + 255) & ~(size_t)255;
        return r;
    };
    __bf16* Bw    = (__bf16*)(ws + take((size_t)16 * 8 * 32 * 16 * sizeof(__bf16)));
    __bf16* Brou  = (__bf16*)(ws + take((size_t)4 * 32 * 16 * sizeof(__bf16)));
    float*  scale = (float*) (ws + take((size_t)E * sizeof(float)));
    float*  bvec  = (float*) (ws + take((size_t)E * 16 * sizeof(float)));
    __bf16* Xb    = (__bf16*)(ws + take((size_t)ntiles * 8 * 32 * 16 * sizeof(__bf16)));
    float*  sA    = (float*) (ws + take((size_t)V * 16 * sizeof(float)));
    float*  sB2   = (float*) (ws + take((size_t)V * 16 * sizeof(float)));

    prep_weights<<<(4224 + 255) / 256, 256, 0, stream>>>(xi_w, rou_w, Bw, Brou);
    prep_edges<<<1024, 256, 0, stream>>>(X_Node, X_Neis, dg, emb, Brou, rou_b,
                                         Xb, bvec, scale, ntiles);

    const int nstate = V * 16;
    const int zb = (nstate + 255) / 256;
    const size_t shmem = (size_t)16 * 8 * 32 * 16 * sizeof(__bf16);  // 128 KB

    // step 1: states = segment_sum(b)        (states0 == 0, GEMM skipped)
    zero_kernel<<<zb, 256, 0, stream>>>(sA, nstate);
    prop_step<<<512, 256, shmem, stream>>>(sB2, sA, Xb, Bw, xi_b, bvec, scale,
                                           X_Node, X_Neis, ntiles, 1);
    // step 2: sA -> sB2
    zero_kernel<<<zb, 256, 0, stream>>>(sB2, nstate);
    prop_step<<<512, 256, shmem, stream>>>(sA, sB2, Xb, Bw, xi_b, bvec, scale,
                                           X_Node, X_Neis, ntiles, 0);
    // step 3: sB2 -> sA
    zero_kernel<<<zb, 256, 0, stream>>>(sA, nstate);
    prop_step<<<512, 256, shmem, stream>>>(sB2, sA, Xb, Bw, xi_b, bvec, scale,
                                           X_Node, X_Neis, ntiles, 0);

    final_kernel<<<(V + 255) / 256, 256, 0, stream>>>(emb, sA, lin_w, lin_b,
                                                      (float*)d_out, V);
}
